// QSupFullNet_7043746365619
// MI455X (gfx1250) — matrix operand order repaired and checked by reading, not compiled, not run on hardware
//
#include <hip/hip_runtime.h>
#include <hip/hip_bf16.h>
#include <stdint.h>

// ---------------- problem constants ----------------
#define B_DIM 2048
#define I_DIM 1024
#define H_DIM 2048
#define C_DIM 1000
#define CPAD  1024            // classifier N padded to multiple of 128
#define S_DIM 8
#define N_MAIN (S_DIM * 2 * H_DIM)   // 32768
#define TOPK_K 256
#define EPS_F 1e-8f
#define KSTEP 32              // K slab per LDS buffer (= one WMMA K step)

typedef __bf16 bf16_t;
typedef bf16_t v8bf  __attribute__((ext_vector_type(8)));
typedef bf16_t v16bf __attribute__((ext_vector_type(16)));
typedef float  v8f   __attribute__((ext_vector_type(8)));

#if defined(__has_builtin)
#if __has_builtin(__builtin_amdgcn_tensor_load_to_lds) && \
    __has_builtin(__builtin_amdgcn_s_wait_tensorcnt)
#define USE_TDM 1
#endif
#endif

// ---------------- helpers ----------------
__device__ __forceinline__ uint16_t f32_to_bf16_bits(float f) {
  uint32_t u = __float_as_uint(f);
  uint32_t r = u + 0x7FFFu + ((u >> 16) & 1u);   // round-to-nearest-even
  return (uint16_t)(r >> 16);
}
__device__ __forceinline__ float bf16_bits_to_f32(uint16_t b) {
  return __uint_as_float(((uint32_t)b) << 16);
}

#if defined(USE_TDM)
typedef uint32_t v4u_t __attribute__((ext_vector_type(4)));
typedef int      v4i_t __attribute__((ext_vector_type(4)));
typedef int      v8i_t __attribute__((ext_vector_type(8)));

// LDS byte offset of a __shared__ object: generic (flat) address low 32 bits
// are the LDS offset (flat->LDS mapping truncates to addr[31:0]).
__device__ __forceinline__ uint32_t lds_off_of(const void* p) {
  return (uint32_t)(uintptr_t)p;
}

// Issue one TDM 2D tile load: 128 rows x KSTEP bf16 elems, row stride =
// row_stride_elems, dense into LDS at lds_off. Tracked by TENSORcnt.
// D# per CDNA5 ISA §8.3/8.4: group0 = {flags, lds_addr, global_addr, type=2},
// group1 = {data_size=2B, tensor dims (large: tiles always in-bounds),
//           tile_dim0=KSTEP, tile_dim1=128, tensor_dim0_stride=K}.
__device__ __forceinline__ void tdm_stage(uint32_t lds_off,
                                          const uint16_t* src,
                                          int row_stride_elems) {
  uint64_t ga = (uint64_t)(uintptr_t)src;
  v4u_t g0;
  g0[0] = 1u;                                     // count=1, user mode
  g0[1] = lds_off;                                // lds_addr (bytes)
  g0[2] = (uint32_t)ga;                           // global_addr[31:0]
  g0[3] = ((uint32_t)(ga >> 32) & 0x01FFFFFFu)    // global_addr[56:32]
          | (2u << 30);                           // type = 2 ("image")
  const uint32_t TD = 1u << 30;                   // huge tensor dims: no OOB
  const uint64_t stride = (uint64_t)(uint32_t)row_stride_elems;
  v8i_t g1;
  g1[0] = (int)(1u << 16);                        // data_size=1 (2 bytes)
  g1[1] = (int)((TD & 0xFFFFu) << 16);            // tensor_dim0[15:0]
  g1[2] = (int)(((TD >> 16) & 0xFFFFu)            // tensor_dim0[31:16]
                | ((TD & 0xFFFFu) << 16));        // tensor_dim1[15:0]
  g1[3] = (int)(((TD >> 16) & 0xFFFFu)            // tensor_dim1[31:16]
                | ((uint32_t)KSTEP << 16));       // tile_dim0
  g1[4] = (int)128u;                              // tile_dim1=128, tile_dim2=0
  g1[5] = (int)(uint32_t)stride;                  // tensor_dim0_stride[31:0]
  g1[6] = (int)(uint32_t)((stride >> 32) & 0xFFFFu); // stride[47:32] | d1s=0
  g1[7] = 0;
  v4i_t z4 = {0, 0, 0, 0};
#if __clang_major__ >= 23
  v8i_t z8 = {0, 0, 0, 0, 0, 0, 0, 0};
  __builtin_amdgcn_tensor_load_to_lds(g0, g1, z4, z4, z8, 0);
#else
  __builtin_amdgcn_tensor_load_to_lds(g0, g1, z4, z4, 0);
#endif
}
#endif  // USE_TDM

// Fallback cooperative stage: 256 threads copy a 128 x KSTEP bf16 tile.
__device__ __forceinline__ void coop_stage(uint16_t* dst, const uint16_t* src,
                                           int row_stride_elems) {
  int t = threadIdx.x;           // 0..255
  int row = t >> 1;              // 0..127
  int col = (t & 1) << 4;        // 0 or 16
  *(v16bf*)(dst + row * KSTEP + col) =
      *(const v16bf*)(src + (size_t)row * row_stride_elems + col);
}

// ---------------- conversion kernels ----------------
__global__ void k_f32_to_bf16(const float* __restrict__ src,
                              uint16_t* __restrict__ dst, int n) {
  int i = blockIdx.x * blockDim.x + threadIdx.x;
  if (i < n) dst[i] = f32_to_bf16_bits(src[i]);
}

__global__ void k_clf_to_bf16_pad(const float* __restrict__ src,
                                  uint16_t* __restrict__ dst) {
  int i = blockIdx.x * blockDim.x + threadIdx.x;   // over CPAD*H
  if (i >= CPAD * H_DIM) return;
  int row = i / H_DIM, col = i - row * H_DIM;
  dst[i] = (row < C_DIM) ? f32_to_bf16_bits(src[row * H_DIM + col]) : (uint16_t)0;
}

__global__ void k_trig(const float* __restrict__ ph,
                       float* __restrict__ c, float* __restrict__ s) {
  int i = blockIdx.x * blockDim.x + threadIdx.x;
  if (i < S_DIM * H_DIM) { c[i] = cosf(ph[i]); s[i] = sinf(ph[i]); }
}

// ---------------- WMMA GEMM: D = A(MxK) * B(NxK)^T, bf16 in, f32 acc ----------
// Block = 256 threads = 8 waves, block tile 128x128, wave tile 32x64.
// K slabs (128x32 A tile + 128x32 B tile) staged to LDS by the Tensor Data
// Mover (double-buffered, TENSORcnt-tracked); waves read fragments from LDS:
//   A 16x32 bf16 : lane&15 = row M; K base = (lane>>4)*16; 16 consecutive K
//   B 32x16 bf16 : lane&15 = col N; K base = (lane>>4)*16; 16 consecutive K
//   D 16x16 f32  : lane&15 = col N; rows M = (lane>>4)*8 + r
// EPI 0: main -> exp(-(v+bias)^2) stored bf16 | EPI 1: sigmoid gate, scale
// sup_r/sup_i in place | EPI 2: v+bias -> f32 logits (n < C_DIM).
template <int EPI>
__global__ __launch_bounds__(256) void k_wmma_gemm(
    const uint16_t* __restrict__ A, const uint16_t* __restrict__ Bm,
    int M, int N, int K, const float* __restrict__ bias,
    void* __restrict__ out0, float* __restrict__ out1, float* __restrict__ out2) {
  __shared__ uint16_t shA[2][128 * KSTEP];
  __shared__ uint16_t shB[2][128 * KSTEP];

  const int lane = threadIdx.x & 31;
  const int wid  = threadIdx.x >> 5;
  const int wm   = wid & 3;          // 4 waves along M (32 rows each)
  const int wn   = wid >> 2;         // 2 waves along N (64 cols each)
  const int tile_m0 = blockIdx.y * 128;
  const int tile_n0 = blockIdx.x * 128;
  const int la = lane & 15;
  const int lh = lane >> 4;          // 0 or 1

  v8f acc[2][4];
  const v8f vzero = {0.f, 0.f, 0.f, 0.f, 0.f, 0.f, 0.f, 0.f};
#pragma unroll
  for (int mi = 0; mi < 2; ++mi)
#pragma unroll
    for (int nj = 0; nj < 4; ++nj) acc[mi][nj] = vzero;

  const uint16_t* Ab = A + (size_t)tile_m0 * K;
  const uint16_t* Bb = Bm + (size_t)tile_n0 * K;
  const int nk = K / KSTEP;

#if defined(USE_TDM)
  if (wid == 0) {                       // wave 0 drives the DMA engine
    tdm_stage(lds_off_of(&shA[0][0]), Ab, K);
    tdm_stage(lds_off_of(&shB[0][0]), Bb, K);
  }
#endif

  for (int it = 0; it < nk; ++it) {
    const int cur = it & 1;
#if defined(USE_TDM)
    if (wid == 0) {
      if (it + 1 < nk) {               // kick next slab, drain current pair
        tdm_stage(lds_off_of(&shA[cur ^ 1][0]), Ab + (it + 1) * KSTEP, K);
        tdm_stage(lds_off_of(&shB[cur ^ 1][0]), Bb + (it + 1) * KSTEP, K);
        __builtin_amdgcn_s_wait_tensorcnt(2);
      } else {
        __builtin_amdgcn_s_wait_tensorcnt(0);
      }
    }
#else
    coop_stage(&shA[cur][0], Ab + it * KSTEP, K);
    coop_stage(&shB[cur][0], Bb + it * KSTEP, K);
#endif
    __syncthreads();

    const uint16_t* sa = &shA[cur][0];
    const uint16_t* sb = &shB[cur][0];
    v16bf afrag[2];
#pragma unroll
    for (int mi = 0; mi < 2; ++mi) {
      const uint16_t* ap = sa + (wm * 32 + mi * 16 + la) * KSTEP + lh * 16;
      v8bf lo = *(const v8bf*)ap;            // K..K+7
      v8bf hi = *(const v8bf*)(ap + 8);      // K+8..K+15
      afrag[mi] = __builtin_shufflevector(lo, hi, 0, 1, 2, 3, 4, 5, 6, 7,
                                          8, 9, 10, 11, 12, 13, 14, 15);
    }
    v16bf bfrag[4];
#pragma unroll
    for (int nj = 0; nj < 4; ++nj) {
      const uint16_t* bp = sb + (wn * 64 + nj * 16 + la) * KSTEP + lh * 16;
      bfrag[nj] = *(const v16bf*)bp;         // 16 consecutive K
    }
#pragma unroll
    for (int mi = 0; mi < 2; ++mi)
#pragma unroll
      for (int nj = 0; nj < 4; ++nj)
        acc[mi][nj] = __builtin_amdgcn_wmma_f32_16x16x32_bf16(
            false, afrag[mi], false, bfrag[nj], (short)0, acc[mi][nj],
            false, false);

    __syncthreads();                   // buffer free before DMA overwrites it
  }

  // ---------------- epilogue ----------------
#pragma unroll
  for (int mi = 0; mi < 2; ++mi) {
#pragma unroll
    for (int nj = 0; nj < 4; ++nj) {
      const int n = tile_n0 + wn * 64 + nj * 16 + la;
      const float bv = (EPI == 2 && n >= C_DIM) ? 0.f : bias[n];
#pragma unroll
      for (int r = 0; r < 8; ++r) {
        const int m = tile_m0 + wm * 32 + mi * 16 + lh * 8 + r;
        float v = acc[mi][nj][r] + bv;
        if (EPI == 0) {
          float g = __expf(-v * v);
          ((uint16_t*)out0)[(size_t)m * N + n] = f32_to_bf16_bits(g);
        } else if (EPI == 1) {
          float gate = 1.f / (1.f + __expf(-v));
          size_t idx = (size_t)m * N + n;
          out1[idx] *= gate;
          out2[idx] *= gate;
        } else {
          if (n < C_DIM) ((float*)out0)[(size_t)m * C_DIM + n] = v;
        }
      }
    }
  }
}

// ---------------- norm factor per (b,s): 1/sqrt(sum g^2 + eps) ----------------
__global__ __launch_bounds__(256) void k_norm_factor(
    const uint16_t* __restrict__ g, float* __restrict__ factor) {
  const int bs = blockIdx.x;                       // b*S + s
  const uint16_t* row = g + (size_t)bs * (2 * H_DIM);
  float sum = 0.f;
  for (int i = threadIdx.x; i < 2 * H_DIM; i += 256) {
    float v = bf16_bits_to_f32(row[i]);
    sum += v * v;
  }
  __shared__ float sh[256];
  sh[threadIdx.x] = sum;
  __syncthreads();
  for (int st = 128; st > 0; st >>= 1) {
    if (threadIdx.x < st) sh[threadIdx.x] += sh[threadIdx.x + st];
    __syncthreads();
  }
  if (threadIdx.x == 0) factor[bs] = rsqrtf(sh[0] + EPS_F);
}

// ---------------- superposition over S ----------------
__global__ __launch_bounds__(256) void k_superpose(
    const uint16_t* __restrict__ g, const float* __restrict__ factor,
    const float* __restrict__ cphi, const float* __restrict__ sphi,
    float* __restrict__ sup_r, float* __restrict__ sup_i) {
  int idx = blockIdx.x * blockDim.x + threadIdx.x;   // over B*H
  if (idx >= B_DIM * H_DIM) return;
  int b = idx / H_DIM, h = idx - b * H_DIM;
  float sr = 0.f, si = 0.f;
#pragma unroll
  for (int s = 0; s < S_DIM; ++s) {
    float f  = factor[b * S_DIM + s];
    float a  = bf16_bits_to_f32(g[(size_t)b * N_MAIN + s * 2 * H_DIM + h]) * f;
    float be = bf16_bits_to_f32(g[(size_t)b * N_MAIN + s * 2 * H_DIM + H_DIM + h]) * f;
    float c = cphi[s * H_DIM + h], sp = sphi[s * H_DIM + h];
    sr += a * c - be * sp;
    si += a * sp + be * c;
  }
  sup_r[idx] = sr;
  sup_i[idx] = si;
}

// ---------------- magnitude -> bf16 (gate GEMM input) ----------------
__global__ __launch_bounds__(256) void k_mag(
    const float* __restrict__ sr, const float* __restrict__ si,
    uint16_t* __restrict__ mag_bf) {
  int idx = blockIdx.x * blockDim.x + threadIdx.x;
  if (idx >= B_DIM * H_DIM) return;
  float m = sqrtf(sr[idx] * sr[idx] + si[idx] * si[idx]);
  mag_bf[idx] = f32_to_bf16_bits(m);
}

// ---------------- top-k select + normalized probs (bf16) ----------------
// One block per batch row. Radix threshold search on non-negative float bits.
__global__ __launch_bounds__(256) void k_topk_probs(
    const float* __restrict__ sr, const float* __restrict__ si,
    uint16_t* __restrict__ probs) {
  const int b = blockIdx.x;
  const float* rr = sr + (size_t)b * H_DIM;
  const float* ii = si + (size_t)b * H_DIM;
  float loc[8];
#pragma unroll
  for (int j = 0; j < 8; ++j) {
    int h = threadIdx.x + j * 256;
    float v = rr[h] * rr[h] + ii[h] * ii[h];
    loc[j] = v;
  }
  __shared__ int scnt;
  uint32_t T = 0;
  for (int bit = 31; bit >= 0; --bit) {
    uint32_t cand = T | (1u << bit);
    if (threadIdx.x == 0) scnt = 0;
    __syncthreads();
    int c = 0;
#pragma unroll
    for (int j = 0; j < 8; ++j)
      if (__float_as_uint(loc[j]) >= cand) ++c;
    atomicAdd(&scnt, c);
    __syncthreads();
    if (scnt >= TOPK_K) T = cand;
    __syncthreads();
  }
  // deterministic tree-reduced masked sum
  __shared__ float sh[256];
  float ls = 0.f;
#pragma unroll
  for (int j = 0; j < 8; ++j)
    if (__float_as_uint(loc[j]) >= T) ls += loc[j];
  sh[threadIdx.x] = ls;
  __syncthreads();
  for (int st = 128; st > 0; st >>= 1) {
    if (threadIdx.x < st) sh[threadIdx.x] += sh[threadIdx.x + st];
    __syncthreads();
  }
  float inv = 1.f / (sh[0] + EPS_F);
#pragma unroll
  for (int j = 0; j < 8; ++j) {
    int h = threadIdx.x + j * 256;
    float p = (__float_as_uint(loc[j]) >= T) ? loc[j] * inv : 0.f;
    probs[(size_t)b * H_DIM + h] = f32_to_bf16_bits(p);
  }
}

// ---------------- host-side launch ----------------
extern "C" void kernel_launch(void* const* d_in, const int* in_sizes, int n_in,
                              void* d_out, int out_size, void* d_ws, size_t ws_size,
                              hipStream_t stream) {
  const float* x      = (const float*)d_in[0];
  const float* W      = (const float*)d_in[1];
  const float* bvec   = (const float*)d_in[2];
  const float* phases = (const float*)d_in[3];
  const float* gate_W = (const float*)d_in[4];
  const float* gate_b = (const float*)d_in[5];
  const float* clf_W  = (const float*)d_in[6];
  const float* clf_b  = (const float*)d_in[7];
  float* out = (float*)d_out;

  char* ws = (char*)d_ws;
  size_t off = 0;
  auto walloc = [&](size_t bytes) -> void* {
    void* p = ws + off;
    off += (bytes + 255) & ~(size_t)255;
    return p;
  };
  uint16_t* xb     = (uint16_t*)walloc((size_t)B_DIM * I_DIM * 2);
  uint16_t* Wb     = (uint16_t*)walloc((size_t)N_MAIN * I_DIM * 2);
  uint16_t* gWb    = (uint16_t*)walloc((size_t)H_DIM * H_DIM * 2);
  uint16_t* cWb    = (uint16_t*)walloc((size_t)CPAD * H_DIM * 2);
  uint16_t* gbuf   = (uint16_t*)walloc((size_t)B_DIM * N_MAIN * 2);
  float*    factor = (float*)walloc((size_t)B_DIM * S_DIM * 4);
  float*    sup_r  = (float*)walloc((size_t)B_DIM * H_DIM * 4);
  float*    sup_i  = (float*)walloc((size_t)B_DIM * H_DIM * 4);
  uint16_t* magb   = (uint16_t*)walloc((size_t)B_DIM * H_DIM * 2);
  uint16_t* probsb = (uint16_t*)walloc((size_t)B_DIM * H_DIM * 2);
  float*    cphi   = (float*)walloc((size_t)S_DIM * H_DIM * 4);
  float*    sphi   = (float*)walloc((size_t)S_DIM * H_DIM * 4);

  const int TB = 256;
  // 1. precision conversion to bf16
  {
    int n = B_DIM * I_DIM;
    k_f32_to_bf16<<<(n + TB - 1) / TB, TB, 0, stream>>>(x, xb, n);
  }
  {
    int n = N_MAIN * I_DIM;
    k_f32_to_bf16<<<(n + TB - 1) / TB, TB, 0, stream>>>(W, Wb, n);
  }
  {
    int n = H_DIM * H_DIM;
    k_f32_to_bf16<<<(n + TB - 1) / TB, TB, 0, stream>>>(gate_W, gWb, n);
  }
  {
    int n = CPAD * H_DIM;
    k_clf_to_bf16_pad<<<(n + TB - 1) / TB, TB, 0, stream>>>(clf_W, cWb);
  }
  {
    int n = S_DIM * H_DIM;
    k_trig<<<(n + TB - 1) / TB, TB, 0, stream>>>(phases, cphi, sphi);
  }

  // 2. main projection GEMM + exp(-v^2) epilogue  (M=2048, N=32768, K=1024)
  {
    dim3 grid(N_MAIN / 128, B_DIM / 128);
    k_wmma_gemm<0><<<grid, TB, 0, stream>>>(xb, Wb, B_DIM, N_MAIN, I_DIM,
                                            bvec, (void*)gbuf, nullptr, nullptr);
  }

  // 3. per-(b,s) normalization factor
  k_norm_factor<<<B_DIM * S_DIM, TB, 0, stream>>>(gbuf, factor);

  // 4. superposition over S
  {
    int n = B_DIM * H_DIM;
    k_superpose<<<(n + TB - 1) / TB, TB, 0, stream>>>(gbuf, factor, cphi, sphi,
                                                      sup_r, sup_i);
  }

  // 5. self-modulation loop (2 steps): mag -> gate GEMM (sigmoid, in-place scale)
  for (int step = 0; step < 2; ++step) {
    int n = B_DIM * H_DIM;
    k_mag<<<(n + TB - 1) / TB, TB, 0, stream>>>(sup_r, sup_i, magb);
    dim3 grid(H_DIM / 128, B_DIM / 128);
    k_wmma_gemm<1><<<grid, TB, 0, stream>>>(magb, gWb, B_DIM, H_DIM, H_DIM,
                                            gate_b, nullptr, sup_r, sup_i);
  }

  // 6. top-k mask + prob normalization
  k_topk_probs<<<B_DIM, TB, 0, stream>>>(sup_r, sup_i, probsb);

  // 7. classifier GEMM (M=2048, N=1024 padded, K=2048) -> logits f32
  {
    dim3 grid(CPAD / 128, B_DIM / 128);
    k_wmma_gemm<2><<<grid, TB, 0, stream>>>(probsb, cWb, B_DIM, CPAD, H_DIM,
                                            clf_b, (void*)out, nullptr, nullptr);
  }
  (void)in_sizes; (void)n_in; (void)out_size; (void)ws_size;
}
